// ModulatorBank_55439437857443
// MI455X (gfx1250) — matrix operand order repaired and checked by reading, not compiled, not run on hardware
//
#include <hip/hip_runtime.h>

// ---------------------------------------------------------------------------
// ModulatorBank: per-batch selected 3x3 grouped conv (reflect pad), 2 out ch.
// bf16 WMMA 16x16x32: M=16 pixels, N={u,v}, K=(channel,tap).
// Memory-bound: 536 MB @ 23.3 TB/s ~ 23 us; bf16 WMMA at 2/16 N-utilization
// needs ~39 GFLOP of matrix work -> far under the HBM roof.
// Software-pipelined: global loads (chunk c+1) in flight across the 9 WMMAs
// of chunk c, double-buffered LDS, one barrier per chunk.
// ---------------------------------------------------------------------------

typedef __attribute__((ext_vector_type(16))) __bf16 v16bf;
typedef __attribute__((ext_vector_type(8)))  __bf16 v8bf;
typedef __attribute__((ext_vector_type(8)))  float  v8f;
typedef __attribute__((ext_vector_type(4)))  float  v4f;

#define Bn   32
#define Cn   256
#define Hn   128
#define Wn   128
#define TAPS 9
#define CCH  32                  // channels per LDS chunk
#define NCH  (Cn / CCH)          // 8 chunks
#define COLP 132                 // padded column pitch (cols 0..129 used)
#define LDSB (3 * COLP * CCH)    // bf16 elems per buffer (25,344 B)
#define NBULK (3 * CCH * 32)     // float4 groups per chunk (3072)
#define PT   (NBULK / 256)       // 12 float4 loads per thread per chunk

// XOR-swizzle ch bits[4:3] with col bits[3:2]: spreads the stride-64B
// transpose stores over 4 banks, keeps 8-ch fragment runs contiguous/16B.
__device__ __forceinline__ int swz(int col, int ch) {
  return ch ^ (((col >> 2) & 3) << 3);
}

// ---------------------------------------------------------------------------
// Kernel 1: gather selected u/v kernels -> bf16, layout [b][uv][tap][c].
// ---------------------------------------------------------------------------
__global__ void mb_pack_weights(const float* __restrict__ ku,
                                const float* __restrict__ kv,
                                const int* __restrict__ ui,
                                const int* __restrict__ vi,
                                __bf16* __restrict__ wpack) {
  int idx = blockIdx.x * blockDim.x + threadIdx.x;
  const int total = Bn * 2 * TAPS * Cn;
  if (idx >= total) return;
  int c  = idx % Cn;
  int t  = (idx / Cn) % TAPS;
  int uv = (idx / (Cn * TAPS)) & 1;
  int b  = idx / (Cn * TAPS * 2);
  int view = uv ? vi[b] : ui[b];
  const float* src = uv ? kv : ku;
  float w = src[((size_t)view * Cn + c) * TAPS + t];   // [V][1][C][3][3]
  wpack[idx] = (__bf16)w;
}

// ---------------------------------------------------------------------------
// Kernel 2: one workgroup per (batch, output row); 8 wave32, 16 px per wave.
// ---------------------------------------------------------------------------
__global__ __launch_bounds__(256, 2)
void mb_conv_wmma(const float* __restrict__ x,
                  const __bf16* __restrict__ wpack,
                  float* __restrict__ out) {
  __shared__ __bf16 lds[2][LDSB];

  const int row  = blockIdx.x;
  const int b    = blockIdx.y;
  const int tid  = threadIdx.x;
  const int lane = tid & 31;
  const int wave = tid >> 5;
  const int px0  = wave * 16;
  const int half = lane >> 4;      // 0/1
  const int m    = lane & 15;

  const float* xb = x + (size_t)b * Cn * Hn * Wn;

  int r0 = row - 1; if (r0 < 0) r0 = 1;
  int r2 = row + 1; if (r2 >= Hn) r2 = Hn - 2;

  // per-thread staging decomposition (constant across chunks)
  const int col4 = tid & 31;                 // 4-col group base (coalesced)
  // edge assignment for threads 0..191
  const int e_ri   = tid >> 6;
  const int e_ch   = tid & 31;
  const int e_side = (tid >> 5) & 1;
  const int e_gr   = (e_ri == 0) ? r0 : (e_ri == 1 ? row : r2);

  v4f   bulk[PT];
  float edge = 0.0f;

  // ---- pipeline stages as lambdas -------------------------------------
  auto load_chunk = [&](int c0) {
#pragma unroll
    for (int i = 0; i < PT; ++i) {
      int g  = tid + 256 * i;                // 0..3071
      int ch = (g >> 5) & 31;
      int ri = g >> 10;
      int gr = (ri == 0) ? r0 : (ri == 1 ? row : r2);
      bulk[i] = *(const v4f*)&xb[((size_t)(c0 + ch) * Hn + gr) * Wn + col4 * 4];
    }
    if (tid < 192)
      edge = xb[((size_t)(c0 + e_ch) * Hn + e_gr) * Wn + (e_side ? Wn - 2 : 1)];
  };

  auto store_chunk = [&](__bf16* bp) {
#pragma unroll
    for (int i = 0; i < PT; ++i) {
      int g  = tid + 256 * i;
      int ch = (g >> 5) & 31;
      int ri = g >> 10;
#pragma unroll
      for (int k = 0; k < 4; ++k) {
        int pc = 1 + col4 * 4 + k;           // padded col 1..128
        bp[(ri * COLP + pc) * CCH + swz(pc, ch)] = (__bf16)bulk[i][k];
      }
    }
    if (tid < 192) {
      int pc = e_side ? 129 : 0;             // reflect edges
      bp[(e_ri * COLP + pc) * CCH + swz(pc, e_ch)] = (__bf16)edge;
    }
  };

  // ---- prologue: stage chunk 0 ----------------------------------------
  load_chunk(0);
  store_chunk(lds[0]);
  __syncthreads();

  v8f acc = {};

  for (int cc = 0; cc < NCH; ++cc) {
    const int c0 = cc * CCH;
    const bool more = (cc + 1 < NCH);

    // issue next chunk's global loads; they stay in flight across the WMMAs
    if (more) load_chunk(c0 + CCH);
    // pull chunk cc+2 toward L2 while HBM pipe is busy
    if (cc + 2 < NCH && tid < 96) {
      int ch = tid & 31, ri = tid >> 5;
      int gr = (ri == 0) ? r0 : (ri == 1 ? row : r2);
      __builtin_prefetch(&xb[((size_t)(c0 + 2 * CCH + ch) * Hn + gr) * Wn], 0, 1);
    }

    const __bf16* bufc = lds[cc & 1];

    // ---- 9 taps: one v_wmma_f32_16x16x32_bf16 each ----
#pragma unroll
    for (int t = 0; t < TAPS; ++t) {
      const int ky = t / 3, kx = t % 3;
      const int colA = px0 + m + kx;

      // A fragment (16x32 MxK): lane(half,m): j<8 -> K=8h+j ; j>=8 -> K=16+8h+(j-8)
      const __bf16* abase = bufc + (ky * COLP + colA) * CCH;
      v8bf alo = *(const v8bf*)(abase + swz(colA, half * 8));
      v8bf ahi = *(const v8bf*)(abase + swz(colA, 16 + half * 8));
      v16bf a;
#pragma unroll
      for (int j = 0; j < 8; ++j) { a[j] = alo[j]; a[8 + j] = ahi[j]; }

      // B fragment (32x16 KxN): lane col = lane&15, j<8 -> K=8h+j ; j>=8 -> K=16+8h+(j-8).
      v16bf bm = {};
      const int ncol = lane & 15;
      if (ncol < 2) {
        const __bf16* wb = wpack
            + ((size_t)(b * 2 + ncol) * TAPS + t) * Cn + c0 + half * 8;
        v8bf w0 = *(const v8bf*)(wb);
        v8bf w1 = *(const v8bf*)(wb + 16);
#pragma unroll
        for (int j = 0; j < 8; ++j) { bm[j] = w0[j]; bm[8 + j] = w1[j]; }
      }

      acc = __builtin_amdgcn_wmma_f32_16x16x32_bf16(
          false, a, false, bm, (short)0, acc, false, false);
    }

    // drain loads, convert, fill the other buffer
    if (more) store_chunk(lds[(cc + 1) & 1]);
    __syncthreads();
  }

  // ---- writeback: D col N=lane&15 (0=u,1=v), rows M = r + 8*half ----
  const int ncol = lane & 15;
  if (ncol < 2) {
    float* o = out + (((size_t)b * 2 + ncol) * Hn + row) * Wn + px0 + half * 8;
#pragma unroll
    for (int r = 0; r < 8; ++r) o[r] = acc[r];
  }
}

// ---------------------------------------------------------------------------
extern "C" void kernel_launch(void* const* d_in, const int* in_sizes, int n_in,
                              void* d_out, int out_size, void* d_ws, size_t ws_size,
                              hipStream_t stream) {
  (void)in_sizes; (void)n_in; (void)out_size; (void)ws_size;
  const float* x  = (const float*)d_in[0];
  const float* ku = (const float*)d_in[1];
  const float* kv = (const float*)d_in[2];
  const int*   ui = (const int*)d_in[3];
  const int*   vi = (const int*)d_in[4];
  float* out = (float*)d_out;
  __bf16* wpack = (__bf16*)d_ws;   // 32*2*2304 bf16 = 288 KB scratch

  const int total = Bn * 2 * TAPS * Cn;
  mb_pack_weights<<<(total + 255) / 256, 256, 0, stream>>>(ku, kv, ui, vi, wpack);

  dim3 grid(Hn, Bn);
  mb_conv_wmma<<<grid, 256, 0, stream>>>(x, wpack, out);
}
